// LSTM_2997887172730
// MI455X (gfx1250) — matrix operand order repaired and checked by reading, not compiled, not run on hardware
//
#include <hip/hip_runtime.h>
#include <hip/hip_bf16.h>

#define VOCAB 32000
#define EMB 512
#define HID 512
#define BATCH 16
#define SEQ 128
#define NWG 64           // 32 WGs for layer0 + 32 WGs for layer1 (pipelined)

typedef _Float16 half8  __attribute__((ext_vector_type(8)));
typedef _Float16 v16h   __attribute__((ext_vector_type(16)));
typedef float    v8f    __attribute__((ext_vector_type(8)));

// ---- WMMA operand loaders (16x16x32 f16, contiguous k order shared by A and B) ----
// A: 16x32 (MxK), base = row-major [16 x ld] f16 tile origin.
//    lane L: m=L&15, hi=L>>4 ; elem i: K = hi*16 + i  (16 contiguous halves)
__device__ __forceinline__ v16h loadA(const _Float16* base, int ld, int k0, int lane) {
  const int m = lane & 15, hi = lane >> 4;
  const _Float16* r = base + (size_t)m * ld + k0 + hi * 16;
  half8 lo = *(const half8*)(r);
  half8 hp = *(const half8*)(r + 8);
  v16h a;
#pragma unroll
  for (int i = 0; i < 8; ++i) { a[i] = lo[i]; a[i + 8] = hp[i]; }
  return a;
}
// B: 32x16 (KxN) where B[k][n] = mat[n][k]; base = row n0 of row-major [N x ld].
//    lane L: n=L&15, hi=L>>4 ; elem i: K = hi*16 + i  (16 contiguous halves)
__device__ __forceinline__ v16h loadB(const _Float16* base, int ld, int k0, int lane) {
  const int n = lane & 15, hi = lane >> 4;
  const _Float16* r = base + (size_t)n * ld + k0 + hi * 16;
  half8 lo = *(const half8*)(r);
  half8 hp = *(const half8*)(r + 8);
  v16h b;
#pragma unroll
  for (int i = 0; i < 8; ++i) { b[i] = lo[i]; b[i + 8] = hp[i]; }
  return b;
}
// B tile from an LDS-resident compacted slice; r already = row*512 + hi*16 (+k0)
__device__ __forceinline__ v16h loadB_sh(const _Float16* r) {
  half8 lo = *(const half8*)(r);
  half8 hp = *(const half8*)(r + 8);
  v16h b;
#pragma unroll
  for (int i = 0; i < 8; ++i) { b[i] = lo[i]; b[i + 8] = hp[i]; }
  return b;
}
__device__ __forceinline__ v8f wmma16(v16h a, v16h b, v8f c) {
  return __builtin_amdgcn_wmma_f32_16x16x32_f16(false, a, false, b, (short)0, c, false, false);
}
// Branchless activations: v_exp_f32 (TRANS, co-executes) + single v_rcp_f32.
// tanh(x) = 1 - 2/(exp(2x)+1): saturates to +/-1 at +/-inf, no NaNs for finite x.
__device__ __forceinline__ float fast_sigmoid(float x) {
  return __builtin_amdgcn_rcpf(1.0f + __expf(-x));
}
__device__ __forceinline__ float fast_tanh(float x) {
  return 1.0f - 2.0f * __builtin_amdgcn_rcpf(__expf(2.0f * x) + 1.0f);
}

// ---------------- prep kernels ----------------
__global__ void k_f32_to_f16(const float* __restrict__ s, _Float16* __restrict__ d, int n) {
  for (int i = blockIdx.x * blockDim.x + threadIdx.x; i < n; i += gridDim.x * blockDim.x)
    d[i] = (_Float16)s[i];
}
__global__ void k_bias(const float* __restrict__ a, const float* __restrict__ b,
                       float* __restrict__ o, int n) {
  int i = blockIdx.x * blockDim.x + threadIdx.x;
  if (i < n) o[i] = a[i] + b[i];
}
// xin[t][b][:] = etab[x[b][t]][:]
__global__ void k_gather(const int* __restrict__ x, const _Float16* __restrict__ etab,
                         _Float16* __restrict__ xin) {
  const int b = blockIdx.x >> 7, t = blockIdx.x & 127;
  const int tok = x[b * SEQ + t];
  const uint32_t* src = (const uint32_t*)(etab + (size_t)tok * EMB);
  uint32_t* dst = (uint32_t*)(xin + ((size_t)t * BATCH + b) * EMB);
  dst[threadIdx.x] = src[threadIdx.x];
}

// ---------------- persistent pipelined LSTM recurrence ----------------
__device__ __forceinline__ void grid_barrier(unsigned* cnt, unsigned target) {
  __threadfence();
  __syncthreads();
  if (threadIdx.x == 0)
    __hip_atomic_fetch_add(cnt, 1u, __ATOMIC_RELEASE, __HIP_MEMORY_SCOPE_AGENT);
  if ((threadIdx.x & 31) == 0) {  // one acquire spin per wave
    while (__hip_atomic_load(cnt, __ATOMIC_ACQUIRE, __HIP_MEMORY_SCOPE_AGENT) < target)
      __builtin_amdgcn_s_sleep(1);
  }
  __syncthreads();
}

// WG (layer = blk>>5, wg = blk&31) owns hidden cols [wg*16, wg*16+16).
// LDS weights compacted: wlds[(mtx*64 + g*16 + c)*512 + k],
//   src row = g*512 + wg*16 + c of w_ih/w_hh (mtx 0/1).
// Wave wv: gate tile j = wv>>1 (also the gate index), mtx = wv&1, full K=512.
// Pipeline round r: layer0 computes t=r (r<128); layer1 computes t=r-1 (r>=1).
__global__ __launch_bounds__(256) void k_lstm(
    const _Float16* __restrict__ xin, const _Float16* __restrict__ wih,
    const _Float16* __restrict__ whh, const float* __restrict__ bsum,
    _Float16* __restrict__ h0buf, _Float16* __restrict__ h1buf,  // ping-pong [2][16][512]
    _Float16* __restrict__ hseq,                                 // [128][16][512]
    float* __restrict__ out_hc, unsigned* __restrict__ bar) {
  extern __shared__ _Float16 wlds[];     // 2*64*512 f16 = 128 KB
  __shared__ float glds[4][2][16][16];   // gate partials (ih/hh) for this WG slice
  const int tid = threadIdx.x, lane = tid & 31, wv = tid >> 5;
  const int layer = blockIdx.x >> 5, wg = blockIdx.x & 31;

  {  // stage this WG's weight slice into LDS (once): 8192 x 16B chunks total
    const _Float16* src0 = wih + (size_t)layer * 4 * HID * HID;
    const _Float16* src1 = whh + (size_t)layer * 4 * HID * HID;
    for (int cid = tid; cid < 2 * 64 * 64; cid += 256) {
      const int mtx = cid >> 12, r64 = (cid >> 6) & 63, ck = (cid & 63) * 8;
      const int g = r64 >> 4, c = r64 & 15;
      const _Float16* s = (mtx ? src1 : src0) + ((size_t)(g * HID + wg * 16 + c)) * HID + ck;
      *(half8*)(wlds + ((size_t)(mtx * 64 + r64) * 512 + ck)) = *(const half8*)s;
    }
  }
  // each thread owns gate/cell element (m, c); cell state stays in a register
  const int m = tid >> 4, c = tid & 15;
  const float bi = bsum[layer * 4 * HID + 0 * HID + wg * 16 + c];
  const float bf = bsum[layer * 4 * HID + 1 * HID + wg * 16 + c];
  const float bg = bsum[layer * 4 * HID + 2 * HID + wg * 16 + c];
  const float bo = bsum[layer * 4 * HID + 3 * HID + wg * 16 + c];
  float creg = 0.0f;
  __syncthreads();

  const int j = wv >> 1, mtx = wv & 1;
  const unsigned wbase =
      (unsigned)(((mtx * 64 + j * 16 + (lane & 15)) * 512) + ((lane >> 4) * 16));

  for (int r = 0; r <= SEQ; ++r) {
    const bool active = (layer == 0) ? (r < SEQ) : (r >= 1);
    const int t = (layer == 0) ? r : (r - 1);
    if (active) {
      const int rdslot = (t ^ 1) & 1, wrslot = t & 1;
      const _Float16* inA = (layer == 0) ? (xin + (size_t)t * BATCH * EMB)
                                         : (h0buf + (size_t)wrslot * BATCH * HID);
      const _Float16* hA = (layer == 0) ? (h0buf + (size_t)rdslot * BATCH * HID)
                                        : (h1buf + (size_t)rdslot * BATCH * HID);
      const _Float16* A = mtx ? hA : inA;
      unsigned wo = wbase;
      asm volatile("" : "+v"(wo));  // re-materialize each round: keep ds loads in-loop
      v8f acc = {};
#pragma unroll 4
      for (int k0 = 0; k0 < HID; k0 += 32)
        acc = wmma16(loadA(A, HID, k0, lane), loadB_sh(wlds + wo + k0), acc);

      const int hi = lane >> 4, nl = lane & 15;
#pragma unroll
      for (int rr = 0; rr < 8; ++rr) glds[j][mtx][rr + hi * 8][nl] = acc[rr];
      __syncthreads();

      const float gi = glds[0][0][m][c] + glds[0][1][m][c] + bi;
      const float gf = glds[1][0][m][c] + glds[1][1][m][c] + bf;
      const float gg = glds[2][0][m][c] + glds[2][1][m][c] + bg;
      const float go = glds[3][0][m][c] + glds[3][1][m][c] + bo;
      const float iv = fast_sigmoid(gi), fv = fast_sigmoid(gf);
      const float gv = fast_tanh(gg), ov = fast_sigmoid(go);
      creg = fv * creg + iv * gv;
      const float hn = ov * fast_tanh(creg);
      const int gcol = wg * 16 + c;
      const _Float16 hh = (_Float16)hn;
      if (layer == 0) {
        h0buf[((size_t)wrslot * BATCH + m) * HID + gcol] = hh;
      } else {
        h1buf[((size_t)wrslot * BATCH + m) * HID + gcol] = hh;
        hseq[((size_t)t * BATCH + m) * HID + gcol] = hh;
      }
      if (t == SEQ - 1) out_hc[(size_t)layer * BATCH * HID + m * HID + gcol] = hn;
    }
    grid_barrier(bar, (unsigned)(NWG * (r + 1)));
  }
  out_hc[(size_t)(2 + layer) * BATCH * HID + m * HID + wg * 16 + c] = creg;
}

// ---------------- tied output projection ----------------
// C[2048,32000] = Hseq[2048,512] @ etab^T + fc_b; store to out[b][t][v].
__global__ __launch_bounds__(256) void k_proj(const _Float16* __restrict__ hseq,
                                              const _Float16* __restrict__ etab,
                                              const float* __restrict__ fcb,
                                              float* __restrict__ out) {
  const int lane = threadIdx.x & 31, wv = threadIdx.x >> 5;
  const int nblk = blockIdx.x % (VOCAB / 128), mblk = blockIdx.x / (VOCAB / 128);
  const int m0 = mblk * 128 + (wv >> 2) * 64;
  const int n0 = nblk * 128 + (wv & 3) * 32;
  v8f acc[4][2] = {};
  for (int k0 = 0; k0 < EMB; k0 += 32) {
    v16h bt[2];
#pragma unroll
    for (int nt = 0; nt < 2; ++nt) {
      bt[nt] = loadB(etab + (size_t)(n0 + nt * 16) * EMB, EMB, k0, lane);
      if (k0 + 32 < EMB)
        __builtin_prefetch(etab + (size_t)(n0 + nt * 16 + (lane & 15)) * EMB + k0 + 32, 0, 0);
    }
#pragma unroll
    for (int mt = 0; mt < 4; ++mt) {
      v16h a = loadA(hseq + (size_t)(m0 + mt * 16) * EMB, EMB, k0, lane);
#pragma unroll
      for (int nt = 0; nt < 2; ++nt) acc[mt][nt] = wmma16(a, bt[nt], acc[mt][nt]);
    }
  }
  const int hi = lane >> 4, nl = lane & 15;
#pragma unroll
  for (int nt = 0; nt < 2; ++nt) {
    const int col = n0 + nt * 16 + nl;
    const float bias = fcb[col];
#pragma unroll
    for (int mt = 0; mt < 4; ++mt)
#pragma unroll
      for (int r = 0; r < 8; ++r) {
        const int mm = m0 + mt * 16 + r + hi * 8;                 // mm = t*16 + b
        const size_t orow = (size_t)(mm & 15) * SEQ + (mm >> 4);  // b*128 + t
        out[orow * VOCAB + col] = acc[mt][nt][r] + bias;
      }
  }
}

extern "C" void kernel_launch(void* const* d_in, const int* in_sizes, int n_in,
                              void* d_out, int out_size, void* d_ws, size_t ws_size,
                              hipStream_t stream) {
  const int*   x   = (const int*)d_in[0];
  const float* emb = (const float*)d_in[1];
  const float* wih = (const float*)d_in[2];
  const float* whh = (const float*)d_in[3];
  const float* bih = (const float*)d_in[4];
  const float* bhh = (const float*)d_in[5];
  const float* fcb = (const float*)d_in[6];
  float* out = (float*)d_out;

  char* ws = (char*)d_ws;
  size_t off = 0;
  auto alloc = [&](size_t bytes) {
    char* p = ws + off;
    off = (off + bytes + 255) & ~(size_t)255;
    return p;
  };
  _Float16* etab  = (_Float16*)alloc((size_t)VOCAB * EMB * 2);
  _Float16* wih16 = (_Float16*)alloc((size_t)2 * 4 * HID * HID * 2);
  _Float16* whh16 = (_Float16*)alloc((size_t)2 * 4 * HID * HID * 2);
  float*    bsum  = (float*)   alloc((size_t)2 * 4 * HID * 4);
  _Float16* xin   = (_Float16*)alloc((size_t)SEQ * BATCH * EMB * 2);
  _Float16* hseq  = (_Float16*)alloc((size_t)SEQ * BATCH * HID * 2);
  _Float16* h0buf = (_Float16*)alloc((size_t)2 * BATCH * HID * 2);
  _Float16* h1buf = (_Float16*)alloc((size_t)2 * BATCH * HID * 2);
  unsigned* bar   = (unsigned*)alloc(256);

  hipMemsetAsync(h0buf, 0, (size_t)2 * BATCH * HID * 2, stream);
  hipMemsetAsync(h1buf, 0, (size_t)2 * BATCH * HID * 2, stream);
  hipMemsetAsync(bar, 0, 256, stream);

  k_f32_to_f16<<<512, 256, 0, stream>>>(emb, etab, VOCAB * EMB);
  k_f32_to_f16<<<256, 256, 0, stream>>>(wih, wih16, 2 * 4 * HID * HID);
  k_f32_to_f16<<<256, 256, 0, stream>>>(whh, whh16, 2 * 4 * HID * HID);
  k_bias<<<16, 256, 0, stream>>>(bih, bhh, bsum, 2 * 4 * HID);
  k_gather<<<BATCH * SEQ, 256, 0, stream>>>(x, etab, xin);

  // dynamic LDS: 2 matrices * 64 rows * 512 halves = 128 KB weight slice per WG
  k_lstm<<<NWG, 256, 131072, stream>>>(xin, wih16, whh16, bsum, h0buf, h1buf, hseq,
                                       out + (size_t)BATCH * SEQ * VOCAB, bar);

  k_proj<<<(VOCAB / 128) * (BATCH * SEQ / 128), 256, 0, stream>>>(hseq, etab, fcb, out);
}
